// lstm_11209864643245
// MI455X (gfx1250) — matrix operand order repaired and checked by reading, not compiled, not run on hardware
//
#include <hip/hip_runtime.h>

// Problem constants (match reference)
#define T_DIM   256
#define B_DIM   128
#define DIN_DIM 20
#define HDIM    1024
#define OUT_DIM 1095
#define OUT_PAD 1152              // OUT_DIM rounded up to 128
#define LAYERS  3
#define MROWS   (T_DIM * B_DIM)   // 32768 rows for all GEMMs

typedef __attribute__((ext_vector_type(16))) _Float16 v16h;
typedef __attribute__((ext_vector_type(8)))  _Float16 v8h;
typedef __attribute__((ext_vector_type(8)))  float    v8f;

union Frag16 { v16h v; v8h h[2]; };

// --- CDNA5 async global->LDS copy (16 bytes per lane) + ASYNCcnt waits -----
__device__ __forceinline__ void async_copy_b128(unsigned lds_off, const _Float16* gsrc) {
    asm volatile("global_load_async_to_lds_b128 %0, %1, off"
                 :: "v"(lds_off), "v"((unsigned long long)(uintptr_t)gsrc)
                 : "memory");
}
__device__ __forceinline__ void wait_asynccnt_8() {
    asm volatile("s_wait_asynccnt 0x8" ::: "memory");
}
__device__ __forceinline__ void wait_asynccnt_0() {
    asm volatile("s_wait_asynccnt 0x0" ::: "memory");
}

// ---------------------------------------------------------------------------
// WMMA GEMM: C(MxN) = A(MxK,f16) x Bt(NpadxK,f16)^T [+ bias].
// Block tile 128x128, K-tile 64 (2 WMMA K-steps), double-buffered LDS fed by
// GLOBAL_LOAD_ASYNC_TO_LDS_B128, pipelined with s_wait_asynccnt.
// 8 waves (wave32) arranged 4x2; each wave owns a 32x64 subtile (2x4 frags).
// HALF_OUT: store f16 (U intermediate); else f32 with bias (final dense).
// ---------------------------------------------------------------------------
template<int BM, int BN, int BK, bool HALF_OUT>
__global__ __launch_bounds__(256)
void wmma_gemm_async(const _Float16* __restrict__ A,   // M x K (row-major)
                     const _Float16* __restrict__ Bt,  // Npad x K (row-major; B transposed)
                     const float* __restrict__ bias,
                     void* __restrict__ Cout,
                     int M, int N, int K)
{
    __shared__ _Float16 sA[2][BM * BK];   // [buf][m][k], stride BK halves
    __shared__ _Float16 sB[2][BN * BK];   // [buf][n][k], stride BK halves

    const int tid   = threadIdx.x;
    const int lane  = tid & 31;           // wave32
    const int wave  = tid >> 5;           // 0..7
    const int l15   = lane & 15;
    const int lhalf = lane >> 4;          // 0 or 1
    const int wm    = wave & 3;           // wave row strip (4 x 32 rows)
    const int wn    = wave >> 2;          // wave col strip (2 x 64 cols)

    const int m0 = blockIdx.y * BM;
    const int n0 = blockIdx.x * BN;

    const unsigned sA_base = (unsigned)(uintptr_t)&sA[0][0];
    const unsigned sB_base = (unsigned)(uintptr_t)&sB[0][0];

    v8f acc[2][4] = {};                   // 8 x (16x16 f32) accumulators

    // Per K-tile: BM*BK/8 = 1024 16-byte chunks per matrix, 4 per thread each.
    auto issue_tile = [&](int k0, int buf) {
        #pragma unroll
        for (int p = 0; p < (BM * BK) / (256 * 8); ++p) {
            int lin = p * 256 + tid;
            int m = lin >> 3;             // BK/8 == 8 chunks per row
            int c = (lin & 7) * 8;        // halves
            async_copy_b128(sA_base + (unsigned)(buf * BM * BK + m * BK + c) * 2,
                            A + (size_t)(m0 + m) * K + (k0 + c));
        }
        #pragma unroll
        for (int p = 0; p < (BN * BK) / (256 * 8); ++p) {
            int lin = p * 256 + tid;
            int n = lin >> 3;
            int c = (lin & 7) * 8;
            async_copy_b128(sB_base + (unsigned)(buf * BN * BK + n * BK + c) * 2,
                            Bt + (size_t)(n0 + n) * K + (k0 + c));
        }
    };

    const int NT = K / BK;
    issue_tile(0, 0);

    for (int i = 0; i < NT; ++i) {
        const int buf = i & 1;
        if (i + 1 < NT) {
            issue_tile((i + 1) * BK, buf ^ 1);
            wait_asynccnt_8();            // tile i resident (in-order completion)
        } else {
            wait_asynccnt_0();
        }
        __syncthreads();

        #pragma unroll
        for (int kk = 0; kk < BK / 32; ++kk) {
            // ---- fragments per ISA 7.12.2 layouts ----
            Frag16 af[2];
            #pragma unroll
            for (int mt = 0; mt < 2; ++mt) {
                int m = wm * 32 + mt * 16 + l15;
                const _Float16* ap = &sA[buf][m * BK + kk * 32];
                // lanes 0-15: K=0..7 / 16..23 ; lanes 16-31: K=8..15 / 24..31
                af[mt].h[0] = *(const v8h*)(ap + lhalf * 8);
                af[mt].h[1] = *(const v8h*)(ap + 16 + lhalf * 8);
            }
            Frag16 bfr[4];
            #pragma unroll
            for (int nt = 0; nt < 4; ++nt) {
                int n = wn * 64 + nt * 16 + l15;
                const _Float16* bp = &sB[buf][n * BK + kk * 32];
                bfr[nt].h[0] = *(const v8h*)(bp + lhalf * 8);
                bfr[nt].h[1] = *(const v8h*)(bp + 16 + lhalf * 8);
            }
            #pragma unroll
            for (int mt = 0; mt < 2; ++mt)
                #pragma unroll
                for (int nt = 0; nt < 4; ++nt)
                    acc[mt][nt] = __builtin_amdgcn_wmma_f32_16x16x32_f16(
                        false, af[mt].v, false, bfr[nt].v,
                        (short)0, acc[mt][nt], false, false);
        }
        __syncthreads();                  // protect buf before it is re-filled
    }

    // ---- epilogue: C/D layout: vgpr r -> row = r + 8*lhalf, col = l15 ----
    #pragma unroll
    for (int mt = 0; mt < 2; ++mt) {
        #pragma unroll
        for (int nt = 0; nt < 4; ++nt) {
            int row_base = m0 + wm * 32 + mt * 16 + lhalf * 8;
            int col      = n0 + wn * 64 + nt * 16 + l15;
            if (col < N) {
                if (HALF_OUT) {
                    _Float16* C = (_Float16*)Cout;
                    #pragma unroll
                    for (int r = 0; r < 8; ++r)
                        C[(size_t)(row_base + r) * N + col] = (_Float16)acc[mt][nt][r];
                } else {
                    float* C = (float*)Cout;
                    float bval = bias ? bias[col] : 0.0f;
                    #pragma unroll
                    for (int r = 0; r < 8; ++r)
                        C[(size_t)(row_base + r) * N + col] = acc[mt][nt][r] + bval;
                }
            }
        }
    }
}

// ---------------------------------------------------------------------------
// Weight pre-pass: transpose (K x N, f32) -> (Npad x K, f16), zero-padding
// rows N..Npad-1 so the GEMM needs no B-side guards.
// ---------------------------------------------------------------------------
__global__ __launch_bounds__(256)
void transpose_w_kernel(const float* __restrict__ src, _Float16* __restrict__ dst,
                        int K, int N, int Npad)
{
    __shared__ float tile[32][33];
    const int k0 = blockIdx.y * 32;
    const int n0 = blockIdx.x * 32;
    #pragma unroll
    for (int i = threadIdx.y; i < 32; i += 8) {
        int k = k0 + i, n = n0 + threadIdx.x;
        tile[i][threadIdx.x] = (k < K && n < N) ? src[(size_t)k * N + n] : 0.0f;
    }
    __syncthreads();
    #pragma unroll
    for (int i = threadIdx.y; i < 32; i += 8) {
        int n = n0 + i, k = k0 + threadIdx.x;
        if (n < Npad && k < K)
            dst[(size_t)n * K + k] = (_Float16)tile[threadIdx.x][i];
    }
}

// ---------------------------------------------------------------------------
// Dense layer 1: h = x @ W1 + b1 (K=20, trivial) -> f16 activations.
// ---------------------------------------------------------------------------
__global__ __launch_bounds__(256)
void dense1_kernel(const float* __restrict__ x, const float* __restrict__ W1,
                   const float* __restrict__ b1, _Float16* __restrict__ h)
{
    __shared__ float sx[DIN_DIM];
    const int row = blockIdx.y;
    const int j   = blockIdx.x * 256 + threadIdx.x;
    if (threadIdx.x < DIN_DIM)
        sx[threadIdx.x] = x[(size_t)row * DIN_DIM + threadIdx.x];
    __syncthreads();
    float acc = b1[j];
    #pragma unroll
    for (int k = 0; k < DIN_DIM; ++k)
        acc = fmaf(sx[k], W1[k * HDIM + j], acc);
    h[(size_t)row * HDIM + j] = (_Float16)acc;
}

// ---------------------------------------------------------------------------
// SRU recurrence: sequential over T, parallel over B*H (131072 threads).
// f16 U / h streams, f32 cell state in registers, f32 c_final output.
// ---------------------------------------------------------------------------
__global__ __launch_bounds__(256)
void sru_scan_kernel(const _Float16* __restrict__ U, const _Float16* __restrict__ hin,
                     const float* __restrict__ bf, const float* __restrict__ br,
                     _Float16* __restrict__ hout, float* __restrict__ c_final)
{
    const int idx  = blockIdx.x * 256 + threadIdx.x;   // 0 .. B*H-1
    const int hidx = idx & (HDIM - 1);
    const int b    = idx >> 10;                        // HDIM == 1024
    const float bfv = bf[hidx];
    const float brv = br[hidx];

    float c = 0.0f;
    const size_t uStep = (size_t)B_DIM * 3 * HDIM;
    const size_t hStep = (size_t)B_DIM * HDIM;
    size_t uBase = (size_t)b * 3 * HDIM + hidx;
    size_t hBase = (size_t)b * HDIM + hidx;

    for (int t = 0; t < T_DIM; ++t) {
        float z  = (float)U[uBase];
        float fp = (float)U[uBase + HDIM];
        float rp = (float)U[uBase + 2 * HDIM];
        float xv = (float)hin[hBase];
        float f = 1.0f / (1.0f + __expf(-(fp + bfv)));
        float r = 1.0f / (1.0f + __expf(-(rp + brv)));
        c = f * c + (1.0f - f) * z;
        hout[hBase] = (_Float16)(r * c + (1.0f - r) * xv);
        uBase += uStep;
        hBase += hStep;
    }
    c_final[idx] = c;
}

// ---------------------------------------------------------------------------
extern "C" void kernel_launch(void* const* d_in, const int* in_sizes, int n_in,
                              void* d_out, int out_size, void* d_ws, size_t ws_size,
                              hipStream_t stream)
{
    (void)in_sizes; (void)n_in; (void)out_size; (void)ws_size;

    const float* x    = (const float*)d_in[0];
    // d_in[1] = c (accepted but unused, as in reference)
    const float* W1   = (const float*)d_in[2];
    const float* b1   = (const float*)d_in[3];
    const float* Wsru = (const float*)d_in[4];   // (L, H, 3H)
    const float* bfS  = (const float*)d_in[5];   // (L, H)
    const float* brS  = (const float*)d_in[6];   // (L, H)
    const float* W3   = (const float*)d_in[7];   // (H, OUT)
    const float* b3   = (const float*)d_in[8];   // (OUT,)

    float* out  = (float*)d_out;                           // (T*B, OUT) f32
    float* cfin = out + (size_t)MROWS * OUT_DIM;           // (L, B, H) f32

    // Workspace (f16): hA | hB | U | Wt_sru (L,3H,H) | Wt3 (OUT_PAD,H)
    _Float16* hA  = (_Float16*)d_ws;
    _Float16* hB  = hA + (size_t)MROWS * HDIM;
    _Float16* U   = hB + (size_t)MROWS * HDIM;
    _Float16* Wt  = U  + (size_t)MROWS * 3 * HDIM;
    _Float16* Wt3 = Wt + (size_t)LAYERS * 3 * HDIM * HDIM;

    // Weight pre-pass: transpose + f32->f16 (zero-padded for dense3)
    for (int l = 0; l < LAYERS; ++l)
        transpose_w_kernel<<<dim3((3 * HDIM) / 32, HDIM / 32), dim3(32, 8), 0, stream>>>(
            Wsru + (size_t)l * HDIM * 3 * HDIM, Wt + (size_t)l * 3 * HDIM * HDIM,
            HDIM, 3 * HDIM, 3 * HDIM);
    transpose_w_kernel<<<dim3(OUT_PAD / 32, HDIM / 32), dim3(32, 8), 0, stream>>>(
        W3, Wt3, HDIM, OUT_DIM, OUT_PAD);

    // Dense layer 1 -> f16 activations
    dense1_kernel<<<dim3(HDIM / 256, MROWS), 256, 0, stream>>>(x, W1, b1, hA);

    // 3 SRU layers: async-fed WMMA GEMM (32768x3072x1024) + fused scan
    _Float16* hcur = hA;
    _Float16* hnxt = hB;
    for (int l = 0; l < LAYERS; ++l) {
        wmma_gemm_async<128, 128, 64, true>
            <<<dim3((3 * HDIM) / 128, MROWS / 128), 256, 0, stream>>>(
                hcur, Wt + (size_t)l * 3 * HDIM * HDIM, nullptr, U,
                MROWS, 3 * HDIM, HDIM);
        sru_scan_kernel<<<dim3((B_DIM * HDIM) / 256), 256, 0, stream>>>(
            U, hcur, bfS + l * HDIM, brS + l * HDIM,
            hnxt, cfin + (size_t)l * B_DIM * HDIM);
        _Float16* tmp = hcur; hcur = hnxt; hnxt = tmp;
    }

    // Dense layer 3 (f32 out, fused bias; B rows zero-padded to OUT_PAD)
    wmma_gemm_async<128, 128, 64, false>
        <<<dim3(OUT_PAD / 128, MROWS / 128), 256, 0, stream>>>(
            hcur, Wt3, b3, out, MROWS, OUT_DIM, HDIM);
}
